// CPUExpertMLP_17454747091080
// MI455X (gfx1250) — matrix operand order repaired and checked by reading, not compiled, not run on hardware
//
#include <hip/hip_runtime.h>

// Problem constants (from the reference).
#define T_TOK  8192
#define DMODEL 2048
#define HDIM   1408
#define NEXP   8
#define TOPK   2
#define ACAP   (T_TOK * TOPK)   // per-expert list capacity (worst case)

// Tiling.
#define ROWS   48               // assignments (rows) per block: 3 WMMA m-tiles
#define MT     3                // ROWS / 16
#define HP     (HDIM + 8)       // LDS pitch (bf16 elems), keeps 16B alignment

typedef __bf16 bf16_t;
typedef __attribute__((ext_vector_type(16))) __bf16 v16bf;
typedef __attribute__((ext_vector_type(8)))  __bf16 v8bf;
typedef __attribute__((ext_vector_type(4)))  __bf16 v4bf;
typedef __attribute__((ext_vector_type(8)))  float  v8f;

// ---------------------------------------------------------------------------
// Load 16 bf16 from two 16B chunks -> WMMA operand vector.
// ---------------------------------------------------------------------------
static __device__ __forceinline__ v16bf ld16x2(const bf16_t* p0, const bf16_t* p1) {
  v16bf r;
  ((v8bf*)&r)[0] = *(const v8bf*)p0;
  ((v8bf*)&r)[1] = *(const v8bf*)p1;
  return r;
}

static __device__ __forceinline__ v8f wmma_bf16(v16bf a, v16bf b, v8f c) {
  return __builtin_amdgcn_wmma_f32_16x16x32_bf16(
      /*neg_a=*/false, a, /*neg_b=*/false, b,
      /*c_mod=*/(short)0, c, /*reuse_a=*/false, /*reuse_b=*/false);
}

// silu via fast v_rcp_f32 (result is rounded to bf16 anyway).
static __device__ __forceinline__ float silu_f(float g) {
  return g * __builtin_amdgcn_rcpf(1.0f + __expf(-g));
}

// ---------------------------------------------------------------------------
// f32 -> bf16 bulk convert (n divisible by 4).
// ---------------------------------------------------------------------------
__global__ void k_cvt_bf16(const float* __restrict__ s, bf16_t* __restrict__ d, int n) {
  int i = (blockIdx.x * blockDim.x + threadIdx.x) * 4;
  if (i >= n) return;
  float4 v = *reinterpret_cast<const float4*>(s + i);
  v4bf o;
  o[0] = (bf16_t)v.x; o[1] = (bf16_t)v.y; o[2] = (bf16_t)v.z; o[3] = (bf16_t)v.w;
  *reinterpret_cast<v4bf*>(d + i) = o;
}

// ---------------------------------------------------------------------------
// Routing: scatter (token, weight) into per-expert lists.
// ---------------------------------------------------------------------------
__global__ void k_route(const int* __restrict__ idx, const float* __restrict__ wts,
                        int* __restrict__ cnt, int* __restrict__ tok,
                        float* __restrict__ wv) {
  int i = blockIdx.x * blockDim.x + threadIdx.x;
  if (i >= T_TOK * TOPK) return;
  int e = idx[i];
  if ((unsigned)e >= NEXP) return;
  int pos = atomicAdd(&cnt[e], 1);
  tok[e * ACAP + pos] = i / TOPK;
  wv [e * ACAP + pos] = wts[i];
}

// ---------------------------------------------------------------------------
// Fused expert MLP: block = (expert e, tile m of 48 assignments).
// Stage 1: h = silu(x Wg^T) * (x Wu^T)  -> LDS slab [ROWS x HDIM] bf16
// Stage 2: y += w * (h Wd^T)            -> global f32 atomics
// Waves register-block all 3 m-tiles so every B load feeds 3 WMMAs.
// ---------------------------------------------------------------------------
__global__ void __launch_bounds__(256, 1)
k_moe(const bf16_t* __restrict__ xb,  const bf16_t* __restrict__ wgb,
      const bf16_t* __restrict__ wub, const bf16_t* __restrict__ wdb,
      const int* __restrict__ cnt, const int* __restrict__ tok,
      const float* __restrict__ wv, float* __restrict__ y) {
  __shared__ bf16_t hsh[ROWS * HP];        // ~136 KB
  __shared__ int    tokS[ROWS];
  __shared__ float  wS[ROWS];

  const int e = blockIdx.x % NEXP;
  const int m = blockIdx.x / NEXP;
  const int cntv = cnt[e];
  const int pos0 = m * ROWS;
  if (pos0 >= cntv) return;

  const int tid  = threadIdx.x;
  const int wid  = tid >> 5;
  const int lane = tid & 31;
  const int l    = lane & 15;
  const int hi   = lane >> 4;
  const int koffA = hi * 8;                // A operand K-offset (ISA A layout)
  const int koffB = hi * 16;               // contiguous K-offset (unused; B loads use koffA)

  if (tid < ROWS) {
    int p = pos0 + tid;
    if (p < cntv) { tokS[tid] = tok[e * ACAP + p]; wS[tid] = wv[e * ACAP + p]; }
    else          { tokS[tid] = 0;                 wS[tid] = 0.0f; }
  }
  __syncthreads();

  // ---------------- Stage 1: gate/up GEMMs + SwiGLU into LDS ----------------
  {
    const bf16_t* xr0 = xb + (size_t)tokS[ 0 + l] * DMODEL;
    const bf16_t* xr1 = xb + (size_t)tokS[16 + l] * DMODEL;
    const bf16_t* xr2 = xb + (size_t)tokS[32 + l] * DMODEL;

    for (int nt = wid; nt < HDIM / 16; nt += 8) {
      const bf16_t* grow = wgb + ((size_t)e * HDIM + nt * 16 + l) * DMODEL;
      const bf16_t* urow = wub + ((size_t)e * HDIM + nt * 16 + l) * DMODEL;
      // Prefetch next n-tile's weight rows (global_prefetch_b8) to hide the
      // L2 latency at the tile transition; weight streams are L2-resident.
      if (nt + 8 < HDIM / 16) {
        __builtin_prefetch(grow + (size_t)8 * 16 * DMODEL, 0, 3);
        __builtin_prefetch(urow + (size_t)8 * 16 * DMODEL, 0, 3);
      }
      v8f cg0 = {}, cu0 = {}, cg1 = {}, cu1 = {}, cg2 = {}, cu2 = {};
#pragma unroll 2
      for (int k = 0; k < DMODEL; k += 32) {
        v16bf a0 = ld16x2(xr0 + k + koffA, xr0 + k + koffA + 16);
        v16bf a1 = ld16x2(xr1 + k + koffA, xr1 + k + koffA + 16);
        v16bf a2 = ld16x2(xr2 + k + koffA, xr2 + k + koffA + 16);
        v16bf bg = ld16x2(grow + k + koffA, grow + k + koffA + 16);
        v16bf bu = ld16x2(urow + k + koffA, urow + k + koffA + 16);
        cg0 = wmma_bf16(a0, bg, cg0);  cu0 = wmma_bf16(a0, bu, cu0);
        cg1 = wmma_bf16(a1, bg, cg1);  cu1 = wmma_bf16(a1, bu, cu1);
        cg2 = wmma_bf16(a2, bg, cg2);  cu2 = wmma_bf16(a2, bu, cu2);
      }
      const int col = nt * 16 + l;
#pragma unroll
      for (int r = 0; r < 8; ++r) {
        const int rr = r + 8 * hi;
        hsh[(size_t)( 0 + rr) * HP + col] = (bf16_t)(silu_f(cg0[r]) * cu0[r]);
        hsh[(size_t)(16 + rr) * HP + col] = (bf16_t)(silu_f(cg1[r]) * cu1[r]);
        hsh[(size_t)(32 + rr) * HP + col] = (bf16_t)(silu_f(cg2[r]) * cu2[r]);
      }
    }
  }
  __syncthreads();

  // ---------------- Stage 2: down-proj GEMM + weighted scatter --------------
  {
    const bf16_t* h0 = &hsh[(size_t)( 0 + l) * HP];
    const bf16_t* h1 = &hsh[(size_t)(16 + l) * HP];
    const bf16_t* h2 = &hsh[(size_t)(32 + l) * HP];

    // Hoist per-row output base offsets and routing weights out of the dt loop.
    int   ybase[24];
    float wreg[24];
#pragma unroll
    for (int r = 0; r < 8; ++r) {
      const int rr = r + 8 * hi;
      ybase[r]      = tokS[ 0 + rr] * DMODEL;
      ybase[8 + r]  = tokS[16 + rr] * DMODEL;
      ybase[16 + r] = tokS[32 + rr] * DMODEL;
      wreg[r]       = wS[ 0 + rr];
      wreg[8 + r]   = wS[16 + rr];
      wreg[16 + r]  = wS[32 + rr];
    }

    for (int dt = wid; dt < DMODEL / 16; dt += 8) {
      const bf16_t* drow = wdb + ((size_t)e * DMODEL + dt * 16 + l) * HDIM;
      if (dt + 8 < DMODEL / 16) {
        __builtin_prefetch(drow + (size_t)8 * 16 * HDIM, 0, 3);
      }
      v8f c0 = {}, c1 = {}, c2 = {};
#pragma unroll 2
      for (int k = 0; k < HDIM; k += 32) {
        v16bf b  = ld16x2(drow + k + koffA, drow + k + koffA + 16);
        v16bf a0 = ld16x2(h0 + k + koffA, h0 + k + koffA + 16);
        v16bf a1 = ld16x2(h1 + k + koffA, h1 + k + koffA + 16);
        v16bf a2 = ld16x2(h2 + k + koffA, h2 + k + koffA + 16);
        c0 = wmma_bf16(a0, b, c0);
        c1 = wmma_bf16(a1, b, c1);
        c2 = wmma_bf16(a2, b, c2);
      }
      const int col = dt * 16 + l;
#pragma unroll
      for (int r = 0; r < 8; ++r) {
        atomicAdd(&y[(size_t)ybase[r]      + col], c0[r] * wreg[r]);
        atomicAdd(&y[(size_t)ybase[8 + r]  + col], c1[r] * wreg[8 + r]);
        atomicAdd(&y[(size_t)ybase[16 + r] + col], c2[r] * wreg[16 + r]);
      }
    }
  }
}

// ---------------------------------------------------------------------------
// Host launcher.
// ---------------------------------------------------------------------------
extern "C" void kernel_launch(void* const* d_in, const int* in_sizes, int n_in,
                              void* d_out, int out_size, void* d_ws, size_t ws_size,
                              hipStream_t stream) {
  const float* x   = (const float*)d_in[0];
  const float* wts = (const float*)d_in[1];
  const float* Wg  = (const float*)d_in[2];
  const float* Wu  = (const float*)d_in[3];
  const float* Wd  = (const float*)d_in[4];
  const int*   idx = (const int*)  d_in[5];
  float* y = (float*)d_out;

  const size_t nX = (size_t)T_TOK * DMODEL;          // 16,777,216
  const size_t nW = (size_t)NEXP * HDIM * DMODEL;    // 23,068,672
  auto align256 = [](size_t v) { return (v + 255) & ~(size_t)255; };

  char* p = (char*)d_ws;
  bf16_t* xb  = (bf16_t*)p; p += align256(nX * sizeof(bf16_t));
  bf16_t* wgb = (bf16_t*)p; p += align256(nW * sizeof(bf16_t));
  bf16_t* wub = (bf16_t*)p; p += align256(nW * sizeof(bf16_t));
  bf16_t* wdb = (bf16_t*)p; p += align256(nW * sizeof(bf16_t));
  int*    cnt = (int*)p;    p += align256(NEXP * sizeof(int));
  int*    tok = (int*)p;    p += align256((size_t)NEXP * ACAP * sizeof(int));
  float*  wvb = (float*)p;  p += align256((size_t)NEXP * ACAP * sizeof(float));

  // Zero output (we accumulate into it) and routing counters.
  hipMemsetAsync(y, 0, (size_t)out_size * sizeof(float), stream);
  hipMemsetAsync(cnt, 0, NEXP * sizeof(int), stream);

  // f32 -> bf16 conversions (weights become L2-resident at 138 MB total).
  {
    const int thr = 256;
    int bx = (int)((nX / 4 + thr - 1) / thr);
    k_cvt_bf16<<<bx, thr, 0, stream>>>(x, xb, (int)nX);
    int bw = (int)((nW / 4 + thr - 1) / thr);
    k_cvt_bf16<<<bw, thr, 0, stream>>>(Wg, wgb, (int)nW);
    k_cvt_bf16<<<bw, thr, 0, stream>>>(Wu, wub, (int)nW);
    k_cvt_bf16<<<bw, thr, 0, stream>>>(Wd, wdb, (int)nW);
  }

  // Routing scatter.
  k_route<<<(T_TOK * TOPK + 255) / 256, 256, 0, stream>>>(idx, wts, cnt, tok, wvb);

  // Fused expert MLP.
  const int tilesM = (T_TOK * TOPK + ROWS - 1) / ROWS;   // 342
  k_moe<<<NEXP * tilesM, 256, 0, stream>>>(xb, wgb, wub, wdb, cnt, tok, wvb, y);
}
